// PAM_28587302322762
// MI455X (gfx1250) — matrix operand order repaired and checked by reading, not compiled, not run on hardware
//
#include <hip/hip_runtime.h>

// Problem constants (from the reference): B=4, C=512, N=4096, D=64
#define B_    4
#define CCH   512
#define NN    4096
#define DD    64
#define MT    16      // query tile (WMMA M)
#define KT    32      // key tile (WMMA K for bf16)
#define WAVES 4
#define CPW   128     // channels per wave (8 x 16-wide WMMA tiles)

#define A_ELEMS ((size_t)B_ * CCH * NN)   // 8,388,608
#define B_ELEMS ((size_t)B_ * NN * DD)    // 1,048,576
#define C_ELEMS ((size_t)B_ * NN * DD)    // 1,048,576

typedef __attribute__((ext_vector_type(16))) __bf16 v16bf;
typedef __attribute__((ext_vector_type(8)))  __bf16 v8bf;
typedef __attribute__((ext_vector_type(4)))  __bf16 v4bf;
typedef __attribute__((ext_vector_type(8)))  float  v8f;

static __device__ __forceinline__ __bf16 f2bf(float f) {
    // round-to-nearest-even f32 -> bf16
    unsigned u = __builtin_bit_cast(unsigned, f);
    unsigned r = u + 0x7FFFu + ((u >> 16) & 1u);
    unsigned short h = (unsigned short)(r >> 16);
    return __builtin_bit_cast(__bf16, h);
}

static __device__ __forceinline__ float rmax16(float x) {
    x = fmaxf(x, __shfl_xor(x, 1, 32));
    x = fmaxf(x, __shfl_xor(x, 2, 32));
    x = fmaxf(x, __shfl_xor(x, 4, 32));
    x = fmaxf(x, __shfl_xor(x, 8, 32));
    return x;
}
static __device__ __forceinline__ float rsum16(float x) {
    x += __shfl_xor(x, 1, 32);
    x += __shfl_xor(x, 2, 32);
    x += __shfl_xor(x, 4, 32);
    x += __shfl_xor(x, 8, 32);
    return x;
}

// contiguous 16 f32 -> v16bf (fallback path only)
static __device__ __forceinline__ v16bf cvt16(const float* __restrict__ p) {
    v16bf r;
    #pragma unroll
    for (int i = 0; i < 16; ++i) r[i] = f2bf(p[i]);
    return r;
}

// ---------------- one-time f32 -> bf16 conversion pass ----------------
__global__ __launch_bounds__(256) void cvt_f32_bf16(const float* __restrict__ src,
                                                    __bf16* __restrict__ dst,
                                                    unsigned n4) {
    unsigned t = blockIdx.x * blockDim.x + threadIdx.x;
    if (t < n4) {
        unsigned i = t * 4u;
        float4 f = *(const float4*)(src + i);
        v4bf o;
        o[0] = f2bf(f.x); o[1] = f2bf(f.y); o[2] = f2bf(f.z); o[3] = f2bf(f.w);
        *(v4bf*)(dst + i) = o;
    }
}

// ---------------- fused flash attention + residual ----------------
// PRE = true : read pre-converted bf16 copies (no conversion VALU in hot loop)
// PRE = false: fallback, convert f32 inline (used only if d_ws is too small)
template <bool PRE>
__global__ __launch_bounds__(128) void flash_attn_residual(
    const float*  __restrict__ a,    // [B, C, N]  values(T) + residual
    const float*  __restrict__ b,    // [B, N, D]  queries
    const float*  __restrict__ c,    // [B, N, D]  keys
    const __bf16* __restrict__ aBF,
    const __bf16* __restrict__ bBF,
    const __bf16* __restrict__ cBF,
    float* __restrict__ out)         // [B, C, N]
{
    const int lane = threadIdx.x & 31;
    const int wave = threadIdx.x >> 5;
    const int half = lane >> 4;     // 0: lanes 0-15, 1: lanes 16-31
    const int l16  = lane & 15;
    const int bat  = blockIdx.y;
    const int m0   = blockIdx.x * MT;
    const int cbase = wave * CPW;

    // P tile, 16 rows x 32 k, stored in natural k order (column = k)
    // so each reader lane's A-fragment is one contiguous 32-byte LDS load.
    __shared__ __bf16 sP[MT][KT];   // row stride 64B -> 16B-aligned subloads

    // ---- Q fragments (A-matrix 16x32 bf16 layout), two d-halves, loaded once ----
    v16bf aQ[2];
    #pragma unroll
    for (int h = 0; h < 2; ++h) {
        if constexpr (PRE) {
            const __bf16* q = bBF + ((size_t)bat * NN + (size_t)(m0 + l16)) * DD + h * 32;
            aQ[h] = *(const v16bf*)(q + half * 16);        // k = half*16 + 0..15
        } else {
            const float* q = b + ((size_t)bat * NN + (size_t)(m0 + l16)) * DD + h * 32;
            #pragma unroll
            for (int v = 0; v < 8; ++v) {
                int kd = half * 16 + 2 * v;
                aQ[h][2 * v]     = f2bf(q[kd]);
                aQ[h][2 * v + 1] = f2bf(q[kd + 1]);
            }
        }
    }

    // ---- output accumulators: 8 tiles of 16(m) x 16(ch) f32 ----
    const v8f vzero = {0.f, 0.f, 0.f, 0.f, 0.f, 0.f, 0.f, 0.f};
    v8f accO[8];
    #pragma unroll
    for (int j = 0; j < 8; ++j) accO[j] = vzero;

    float mrow[8], lrow[8];
    #pragma unroll
    for (int v = 0; v < 8; ++v) { mrow[v] = -3.0e38f; lrow[v] = 0.f; }

    // ================= streaming flash loop over key tiles =================
    for (int n0 = 0; n0 < NN; n0 += KT) {
        // K fragments (B-matrix 32x16 bf16): col = key (lane%16), K = half*16 + 2v(+1)
        v16bf bK[2][2];
        #pragma unroll
        for (int t = 0; t < 2; ++t) {
            #pragma unroll
            for (int h = 0; h < 2; ++h) {
                if constexpr (PRE) {
                    bK[t][h] = *(const v16bf*)(cBF
                        + ((size_t)bat * NN + (size_t)(n0 + t * 16 + l16)) * DD
                        + h * 32 + half * 16);
                } else {
                    bK[t][h] = cvt16(c
                        + ((size_t)bat * NN + (size_t)(n0 + t * 16 + l16)) * DD
                        + h * 32 + half * 16);
                }
            }
        }
        // prefetch next key tile (emits global_prefetch_b8)
        if (n0 + KT < NN) {
            if constexpr (PRE)
                __builtin_prefetch(cBF + ((size_t)bat * NN + (size_t)(n0 + KT + l16)) * DD, 0, 1);
            else
                __builtin_prefetch(c + ((size_t)bat * NN + (size_t)(n0 + KT + l16)) * DD, 0, 1);
        }

        // S = Q * K^T : two 16x16 f32 tiles, contraction d=0..63 in two WMMAs each
        v8f S0 = vzero, S1 = vzero;
        S0 = __builtin_amdgcn_wmma_f32_16x16x32_bf16(false, aQ[0], false, bK[0][0], (short)0, S0, false, false);
        S0 = __builtin_amdgcn_wmma_f32_16x16x32_bf16(false, aQ[1], false, bK[0][1], (short)0, S0, false, false);
        S1 = __builtin_amdgcn_wmma_f32_16x16x32_bf16(false, aQ[0], false, bK[1][0], (short)0, S1, false, false);
        S1 = __builtin_amdgcn_wmma_f32_16x16x32_bf16(false, aQ[1], false, bK[1][1], (short)0, S1, false, false);

        __syncthreads();  // all waves done reading previous sP

        // online softmax update (D-layout: row m = half*8+v lives across 16 lanes)
        // natural write positions: pos(k) = k
        const int pos0 = l16;        // k = l16
        const int pos1 = 16 + l16;   // k = 16 + l16
        float scl[8];
        #pragma unroll
        for (int v = 0; v < 8; ++v) {
            float tm = rmax16(fmaxf(S0[v], S1[v]));
            float nm = fmaxf(mrow[v], tm);
            float sc = __expf(mrow[v] - nm);
            mrow[v] = nm;
            float p0 = __expf(S0[v] - nm);
            float p1 = __expf(S1[v] - nm);
            lrow[v] = lrow[v] * sc + rsum16(p0 + p1);
            scl[v] = sc;
            if (wave == 0) {
                int m = half * 8 + v;
                sP[m][pos0] = f2bf(p0);
                sP[m][pos1] = f2bf(p1);
            }
        }
        __syncthreads();  // sP ready

        // A-fragment of P: one contiguous 32-byte LDS load per lane
        v16bf aP = *(const v16bf*)&sP[l16][half * 16];

        // rescale accumulators and accumulate P * V (V[n, ch] = a[bat, ch, n])
        #pragma unroll
        for (int j = 0; j < 8; ++j) {
            #pragma unroll
            for (int v = 0; v < 8; ++v) accO[j][v] *= scl[v];

            v16bf bV;  // B layout: col = channel, K = key offset (contiguous in a's last dim)
            if constexpr (PRE) {
                bV = *(const v16bf*)(aBF
                    + ((size_t)bat * CCH + (size_t)(cbase + j * 16 + l16)) * (size_t)NN
                    + n0 + half * 16);
            } else {
                bV = cvt16(a
                    + ((size_t)bat * CCH + (size_t)(cbase + j * 16 + l16)) * (size_t)NN
                    + n0 + half * 16);
            }
            accO[j] = __builtin_amdgcn_wmma_f32_16x16x32_bf16(false, aP, false, bV, (short)0, accO[j], false, false);
        }
    }

    // ---- epilogue: normalize by softmax sum, add residual a (f32), store f32 ----
    #pragma unroll
    for (int j = 0; j < 8; ++j) {
        int ch = cbase + j * 16 + l16;
        const float* arow = a   + ((size_t)bat * CCH + ch) * (size_t)NN + m0;
        float*       orow = out + ((size_t)bat * CCH + ch) * (size_t)NN + m0;
        #pragma unroll
        for (int v = 0; v < 8; ++v) {
            int m = half * 8 + v;
            orow[m] = accO[j][v] / lrow[v] + arow[m];
        }
    }
}

extern "C" void kernel_launch(void* const* d_in, const int* in_sizes, int n_in,
                              void* d_out, int out_size, void* d_ws, size_t ws_size,
                              hipStream_t stream) {
    (void)in_sizes; (void)n_in; (void)out_size;
    const float* a = (const float*)d_in[0];   // [B, C, N]
    const float* b = (const float*)d_in[1];   // [B, N, D]
    const float* c = (const float*)d_in[2];   // [B, N, D]
    float* out = (float*)d_out;               // [B, C, N]

    dim3 grid(NN / MT, B_);   // 256 query tiles x 4 batches
    dim3 block(WAVES * 32);   // 4 wave32s

    const size_t needed = (A_ELEMS + B_ELEMS + C_ELEMS) * sizeof(unsigned short);
    if (d_ws != nullptr && ws_size >= needed) {
        __bf16* aBF = (__bf16*)d_ws;
        __bf16* bBF = aBF + A_ELEMS;
        __bf16* cBF = bBF + B_ELEMS;
        // one-time (per launch) f32 -> bf16 conversion of all operands
        {
            unsigned n4 = (unsigned)(A_ELEMS / 4);
            cvt_f32_bf16<<<(n4 + 255) / 256, 256, 0, stream>>>(a, aBF, n4);
            n4 = (unsigned)(B_ELEMS / 4);
            cvt_f32_bf16<<<(n4 + 255) / 256, 256, 0, stream>>>(b, bBF, n4);
            n4 = (unsigned)(C_ELEMS / 4);
            cvt_f32_bf16<<<(n4 + 255) / 256, 256, 0, stream>>>(c, cBF, n4);
        }
        flash_attn_residual<true><<<grid, block, 0, stream>>>(a, b, c, aBF, bBF, cBF, out);
    } else {
        flash_attn_residual<false><<<grid, block, 0, stream>>>(a, b, c,
                                                               nullptr, nullptr, nullptr, out);
    }
}
